// GaitConv_30288109372146
// MI455X (gfx1250) — compile-verified, matrix operand order checked by reading, not run on hardware
//
#include <hip/hip_runtime.h>
#include <hip/hip_bf16.h>

#define N_    100000
#define K_    9
#define NK_   900000
#define NTIL_N   (N_ / 16)    // 6250
#define NTIL_NK  (NK_ / 16)   // 56250

typedef __attribute__((ext_vector_type(16))) _Float16 v16h;
typedef __attribute__((ext_vector_type(8)))  float    v8f;

__device__ __forceinline__ float eluf(float v) {
    return v > 0.f ? v : (__expf(v) - 1.f);
}

// ---------------------------------------------------------------------------
// Prep: zero stats region (768 floats + 8 ints) and convert weights to f16.
// wh layout (f16 idx): 0 lin_w[64x32] | 2048 m1_w1 pad[32x32] | 3072 m1_w2[32x32]
//                      4096 m2_w pad[96x96] | 13312 out_w[160x128]   (33792 total)
// ---------------------------------------------------------------------------
__global__ __launch_bounds__(256) void k_prep(float* stats, _Float16* wh,
        const float* lin_w, const float* m1w1, const float* m1w2,
        const float* m2w, const float* outw) {
    int g = blockIdx.x * 256 + threadIdx.x;
    if (g < 768) stats[g] = 0.f;
    else if (g < 776) ((int*)stats)[g] = 0;
    if (g < 2048) {
        wh[g] = (_Float16)lin_w[g];
    } else if (g < 3072) {
        int j = g - 2048, k = j / 32, n = j % 32;
        wh[g] = (_Float16)(k < 8 ? m1w1[k * 32 + n] : 0.f);
    } else if (g < 4096) {
        wh[g] = (_Float16)m1w2[g - 3072];
    } else if (g < 13312) {
        int j = g - 4096, k = j / 96, n = j % 96;
        wh[g] = (_Float16)((k < 72 && n < 81) ? m2w[k * 81 + n] : 0.f);
    } else if (g < 33792) {
        wh[g] = (_Float16)outw[g - 13312];
    }
}

// ---------------------------------------------------------------------------
// Node GEMM: h_pre = elu(x[100000,64] @ lin_w[64,32] + lin_b), f16 store, bn0 stats
// ---------------------------------------------------------------------------
__global__ __launch_bounds__(256) void k_node_gemm(const float* x, const float* lb,
        const _Float16* w, _Float16* h_pre, float* st_s, float* st_q) {
    __shared__ float ssum[32], ssq[32];
    int tid = threadIdx.x;
    if (tid < 32) { ssum[tid] = 0.f; ssq[tid] = 0.f; }
    __syncthreads();
    int wave = tid >> 5, lane = tid & 31;
    int rt = blockIdx.x * 8 + wave;
    if (rt < NTIL_N) {
        int base = rt * 16;
        int r = lane & 15, kb = (lane & 16) ? 8 : 0;
        v16h a0, a1;
#pragma unroll
        for (int i = 0; i < 8; ++i) {
            const float* xr = x + (size_t)(base + r) * 64;
            a0[i]     = (_Float16)xr[kb + i];
            a0[8 + i] = (_Float16)xr[16 + kb + i];
            a1[i]     = (_Float16)xr[32 + kb + i];
            a1[8 + i] = (_Float16)xr[48 + kb + i];
        }
        for (int ct = 0; ct < 2; ++ct) {
            v16h b0, b1;
#pragma unroll
            for (int i = 0; i < 8; ++i) {
                b0[i]     = w[(kb + i) * 32 + ct * 16 + r];
                b0[8 + i] = w[(16 + kb + i) * 32 + ct * 16 + r];
                b1[i]     = w[(32 + kb + i) * 32 + ct * 16 + r];
                b1[8 + i] = w[(48 + kb + i) * 32 + ct * 16 + r];
            }
            v8f acc = {};
            acc = __builtin_amdgcn_wmma_f32_16x16x32_f16(false, a0, false, b0, (short)0, acc, false, false);
            acc = __builtin_amdgcn_wmma_f32_16x16x32_f16(false, a1, false, b1, (short)0, acc, false, false);
            int col = ct * 16 + r;
            float bias = lb[col];
            float ls = 0.f, lq = 0.f;
#pragma unroll
            for (int v = 0; v < 8; ++v) {
                int M = v + ((lane & 16) ? 8 : 0);
                float val = eluf(acc[v] + bias);
                h_pre[(size_t)(base + M) * 32 + col] = (_Float16)val;
                ls += val; lq += val * val;
            }
            atomicAdd(&ssum[col], ls);
            atomicAdd(&ssq[col], lq);
        }
    }
    __syncthreads();
    if (tid < 32) { atomicAdd(&st_s[tid], ssum[tid]); atomicAdd(&st_q[tid], ssq[tid]); }
}

// ---------------------------------------------------------------------------
// Edge GEMM 1: m1 = elu(ef[900000,8] @ m1_w1[8,32] + b1), K zero-padded to 32
// ---------------------------------------------------------------------------
__global__ __launch_bounds__(256) void k_edge_gemm1(const float* ef, const float* b1,
        const _Float16* w, _Float16* m1, float* st_s, float* st_q) {
    __shared__ float ssum[32], ssq[32];
    int tid = threadIdx.x;
    if (tid < 32) { ssum[tid] = 0.f; ssq[tid] = 0.f; }
    __syncthreads();
    int wave = tid >> 5, lane = tid & 31;
    int rt = blockIdx.x * 8 + wave;
    if (rt < NTIL_NK) {
        int base = rt * 16;
        int r = lane & 15, kb = (lane & 16) ? 8 : 0;
        v16h a;
#pragma unroll
        for (int i = 0; i < 16; ++i) a[i] = (_Float16)0.f;
        if (!(lane & 16)) {
#pragma unroll
            for (int i = 0; i < 8; ++i) a[i] = (_Float16)ef[(size_t)(base + r) * 8 + i];
        }
        for (int ct = 0; ct < 2; ++ct) {
            v16h bf;
#pragma unroll
            for (int i = 0; i < 8; ++i) {
                bf[i]     = w[(kb + i) * 32 + ct * 16 + r];
                bf[8 + i] = w[(16 + kb + i) * 32 + ct * 16 + r];
            }
            v8f acc = {};
            acc = __builtin_amdgcn_wmma_f32_16x16x32_f16(false, a, false, bf, (short)0, acc, false, false);
            int col = ct * 16 + r;
            float bias = b1[col];
            float ls = 0.f, lq = 0.f;
#pragma unroll
            for (int v = 0; v < 8; ++v) {
                int M = v + ((lane & 16) ? 8 : 0);
                float val = eluf(acc[v] + bias);
                m1[(size_t)(base + M) * 32 + col] = (_Float16)val;
                ls += val; lq += val * val;
            }
            atomicAdd(&ssum[col], ls);
            atomicAdd(&ssq[col], lq);
        }
    }
    __syncthreads();
    if (tid < 32) { atomicAdd(&st_s[tid], ssum[tid]); atomicAdd(&st_q[tid], ssq[tid]); }
}

// ---------------------------------------------------------------------------
// Edge GEMM 2: m2 = elu(bn1(m1) @ m1_w2[32,32] + b2), in-place f16, bn2 stats
// ---------------------------------------------------------------------------
__global__ __launch_bounds__(256) void k_edge_gemm2(const float* b2, const _Float16* w,
        const float* s1, const float* t1, _Float16* m, float* st_s, float* st_q) {
    __shared__ float ssum[32], ssq[32];
    int tid = threadIdx.x;
    if (tid < 32) { ssum[tid] = 0.f; ssq[tid] = 0.f; }
    __syncthreads();
    int wave = tid >> 5, lane = tid & 31;
    int rt = blockIdx.x * 8 + wave;
    if (rt < NTIL_NK) {
        int base = rt * 16;
        int r = lane & 15, kb = (lane & 16) ? 8 : 0;
        v16h a;
#pragma unroll
        for (int i = 0; i < 8; ++i) {
            int k0 = kb + i, k1 = 16 + kb + i;
            const _Float16* mr = m + (size_t)(base + r) * 32;
            a[i]     = (_Float16)(s1[k0] * (float)mr[k0] + t1[k0]);
            a[8 + i] = (_Float16)(s1[k1] * (float)mr[k1] + t1[k1]);
        }
        for (int ct = 0; ct < 2; ++ct) {
            v16h bf;
#pragma unroll
            for (int i = 0; i < 8; ++i) {
                bf[i]     = w[(kb + i) * 32 + ct * 16 + r];
                bf[8 + i] = w[(16 + kb + i) * 32 + ct * 16 + r];
            }
            v8f acc = {};
            acc = __builtin_amdgcn_wmma_f32_16x16x32_f16(false, a, false, bf, (short)0, acc, false, false);
            int col = ct * 16 + r;
            float bias = b2[col];
            float ls = 0.f, lq = 0.f;
#pragma unroll
            for (int v = 0; v < 8; ++v) {
                int M = v + ((lane & 16) ? 8 : 0);
                float val = eluf(acc[v] + bias);
                m[(size_t)(base + M) * 32 + col] = (_Float16)val;
                ls += val; lq += val * val;
            }
            atomicAdd(&ssum[col], ls);
            atomicAdd(&ssq[col], lq);
        }
    }
    __syncthreads();
    if (tid < 32) { atomicAdd(&st_s[tid], ssum[tid]); atomicAdd(&st_q[tid], ssq[tid]); }
}

// ---------------------------------------------------------------------------
// e GEMM: e = elu(ef72[N,72] @ m2_w[72,81] + m2_b); padded to K=96, cols=96
// ---------------------------------------------------------------------------
__global__ __launch_bounds__(256) void k_e_gemm(const float* ef, const float* mb,
        const _Float16* w, _Float16* eout, float* st_s, float* st_q) {
    __shared__ float ssum[96], ssq[96];
    int tid = threadIdx.x;
    if (tid < 96) { ssum[tid] = 0.f; ssq[tid] = 0.f; }
    __syncthreads();
    int wave = tid >> 5, lane = tid & 31;
    int rt = blockIdx.x * 8 + wave;
    if (rt < NTIL_N) {
        int base = rt * 16;
        int r = lane & 15, kb = (lane & 16) ? 8 : 0;
        v16h af[3];
#pragma unroll
        for (int kc = 0; kc < 3; ++kc) {
#pragma unroll
            for (int i = 0; i < 8; ++i) {
                int k0 = kc * 32 + kb + i, k1 = kc * 32 + 16 + kb + i;
                af[kc][i]     = (k0 < 72) ? (_Float16)ef[(size_t)(base + r) * 72 + k0] : (_Float16)0.f;
                af[kc][8 + i] = (k1 < 72) ? (_Float16)ef[(size_t)(base + r) * 72 + k1] : (_Float16)0.f;
            }
        }
        for (int ct = 0; ct < 6; ++ct) {
            v8f acc = {};
#pragma unroll
            for (int kc = 0; kc < 3; ++kc) {
                v16h bf;
#pragma unroll
                for (int i = 0; i < 8; ++i) {
                    bf[i]     = w[(kc * 32 + kb + i) * 96 + ct * 16 + r];
                    bf[8 + i] = w[(kc * 32 + 16 + kb + i) * 96 + ct * 16 + r];
                }
                acc = __builtin_amdgcn_wmma_f32_16x16x32_f16(false, af[kc], false, bf, (short)0, acc, false, false);
            }
            int col = ct * 16 + r;
            if (col < 81) {
                float bias = mb[col];
                float ls = 0.f, lq = 0.f;
#pragma unroll
                for (int v = 0; v < 8; ++v) {
                    int M = v + ((lane & 16) ? 8 : 0);
                    float val = eluf(acc[v] + bias);
                    eout[(size_t)(base + M) * 96 + col] = (_Float16)val;
                    ls += val; lq += val * val;
                }
                atomicAdd(&ssum[col], ls);
                atomicAdd(&ssq[col], lq);
            }
        }
    }
    __syncthreads();
    if (tid < 81) { atomicAdd(&st_s[tid], ssum[tid]); atomicAdd(&st_q[tid], ssq[tid]); }
}

// ---------------------------------------------------------------------------
// Per-node grouped conv: eout[n,c,m] = act( sum_k bn(ein)[n,c,k]*w[c,m,k] + b[cm] )
// ---------------------------------------------------------------------------
__global__ __launch_bounds__(256) void k_conv(const _Float16* ein,
        const float* si, const float* ti, const float* w, const float* bias,
        _Float16* eout, float* st_s, float* st_q, int do_elu) {
    __shared__ float ssum[96], ssq[96];
    int tid = threadIdx.x;
    if (tid < 96) { ssum[tid] = 0.f; ssq[tid] = 0.f; }
    __syncthreads();
    long g = (long)blockIdx.x * 256 + tid;
    if (g < (long)N_ * 81) {
        int n = (int)(g / 81), cm = (int)(g % 81);
        int c = cm / 9, mm = cm % 9;
        float acc = bias[cm];
#pragma unroll
        for (int k = 0; k < 9; ++k) {
            int e = c * 9 + k;
            float xv = si[e] * (float)ein[(size_t)n * 96 + e] + ti[e];
            acc += xv * w[c * 81 + mm * 9 + k];
        }
        if (do_elu) acc = eluf(acc);
        eout[(size_t)n * 96 + cm] = (_Float16)acc;
        atomicAdd(&ssum[cm], acc);
        atomicAdd(&ssq[cm], acc * acc);
    }
    __syncthreads();
    if (tid < 81) { atomicAdd(&st_s[tid], ssum[tid]); atomicAdd(&st_q[tid], ssq[tid]); }
}

// ---------------------------------------------------------------------------
// Histogram of edge types (8 bins)
// ---------------------------------------------------------------------------
__global__ __launch_bounds__(256) void k_hist(const int* et, int* hist) {
    __shared__ int h[8];
    int tid = threadIdx.x;
    if (tid < 8) h[tid] = 0;
    __syncthreads();
    long g = (long)blockIdx.x * 256 + tid;
    if (g < NK_) atomicAdd(&h[et[g] & 7], 1);
    __syncthreads();
    if (tid < 8) atomicAdd(&hist[tid], h[tid]);
}

// ---------------------------------------------------------------------------
// BN finalize: scale = g*rsqrt(var+eps), shift = b - mean*scale
// ---------------------------------------------------------------------------
__global__ void k_finalize_bn(const float* ss, const float* sq, const float* g,
        const float* b, float* s, float* t, int nch, float inv) {
    int i = threadIdx.x;
    if (i >= nch) return;
    float mean = ss[i] * inv;
    float var  = sq[i] * inv - mean * mean;
    float sc   = g[i] * rsqrtf(var + 1e-5f);
    s[i] = sc;
    t[i] = b[i] - mean * sc;
}

// Edge-type embedding branch: only 8 distinct rows -> weighted BN via histogram
__global__ void k_finalize_et(const int* hist, const float* emb,
        const float* g, const float* b, float* et_tab) {
    int ch = threadIdx.x;
    if (ch >= 16) return;
    float vals[8], ws = 0.f, wq = 0.f;
    for (int t8 = 0; t8 < 8; ++t8) {
        float v = eluf(emb[t8 * 16 + ch]);
        vals[t8] = v;
        float c = (float)hist[t8];
        ws += c * v; wq += c * v * v;
    }
    float inv = 1.f / (float)NK_;
    float mean = ws * inv, var = wq * inv - mean * mean;
    float sc = g[ch] * rsqrtf(var + 1e-5f);
    float sh = b[ch] - mean * sc;
    for (int t8 = 0; t8 < 8; ++t8) et_tab[t8 * 16 + ch] = vals[t8] * sc + sh;
}

// ---------------------------------------------------------------------------
// Final fused kernel: 16 nodes per block.
// Phase 1 (all 256 threads): gather + concat + T-transform + depthwise -> LDS f16 A tile
// Phase 2 (8 waves): [16x160] @ out_w[160x128] via WMMA, + out_b -> d_out
// ---------------------------------------------------------------------------
__global__ __launch_bounds__(256) void k_final(
        const int* eidx, const int* etyp,
        const _Float16* h_pre, const _Float16* m2, const _Float16* Tbuf,
        const float* s0, const float* t0, const float* s2, const float* t2,
        const float* sC, const float* tC, const float* et_tab,
        const float* dc_w, const float* dc_b,
        const _Float16* w_out, const float* out_b, float* out) {
    __shared__ float    T_lds[16][96];
    __shared__ _Float16 a_tile[16][160];
    int tid = threadIdx.x;
    int n_loc = tid >> 4, s = tid & 15;
    long node  = (long)blockIdx.x * 16 + n_loc;
    long ebase = node * 9;

    // bnC(T) into LDS (written/read by the same 16 lanes; LDS ops in-order per wave)
    for (int i = s; i < 81; i += 16)
        T_lds[n_loc][i] = sC[i] * (float)Tbuf[node * 96 + i] + tC[i];

    // 5 channels per thread (80 channels total per node)
#pragma unroll
    for (int ci = 0; ci < 5; ++ci) {
        int c = s + 16 * ci;
        float row[9];
        if (c < 32) {
#pragma unroll
            for (int k = 0; k < 9; ++k) {
                int j = eidx[NK_ + ebase + k];              // edge_index[1]
                row[k] = s0[c] * (float)h_pre[(size_t)j * 32 + c] + t0[c];
            }
        } else if (c < 48) {
            int cc = c - 32;
#pragma unroll
            for (int k = 0; k < 9; ++k) {
                int t8 = etyp[ebase + k] & 7;
                row[k] = et_tab[t8 * 16 + cc];
            }
        } else {
            int cc = c - 48;
#pragma unroll
            for (int k = 0; k < 9; ++k)
                row[k] = s2[cc] * (float)m2[(size_t)(ebase + k) * 32 + cc] + t2[cc];
        }
        // x_star @ T
        float row2[9];
#pragma unroll
        for (int j = 0; j < 9; ++j) {
            float acc = 0.f;
#pragma unroll
            for (int k = 0; k < 9; ++k) acc += row[k] * T_lds[n_loc][k * 9 + j];
            row2[j] = acc;
        }
        // depthwise conv (MULT=2)
#pragma unroll
        for (int m = 0; m < 2; ++m) {
            float acc = dc_b[c * 2 + m];
#pragma unroll
            for (int k = 0; k < 9; ++k) acc += row2[k] * dc_w[(c * 2 + m) * 9 + k];
            a_tile[n_loc][c * 2 + m] = (_Float16)acc;
        }
    }
    __syncthreads();

    // output GEMM: each wave owns one 16-col tile of the 128 outputs
    int wave = tid >> 5, lane = tid & 31;
    int r = lane & 15, kb = (lane & 16) ? 8 : 0;
    int ct = wave;
    v8f acc = {};
#pragma unroll
    for (int kc = 0; kc < 5; ++kc) {
        v16h a, bf;
#pragma unroll
        for (int i = 0; i < 8; ++i) {
            a[i]      = a_tile[r][kc * 32 + kb + i];
            a[8 + i]  = a_tile[r][kc * 32 + 16 + kb + i];
            bf[i]     = w_out[(kc * 32 + kb + i) * 128 + ct * 16 + r];
            bf[8 + i] = w_out[(kc * 32 + 16 + kb + i) * 128 + ct * 16 + r];
        }
        acc = __builtin_amdgcn_wmma_f32_16x16x32_f16(false, a, false, bf, (short)0, acc, false, false);
    }
    int col = ct * 16 + r;
    float bias = out_b[col];
#pragma unroll
    for (int v = 0; v < 8; ++v) {
        int M = v + ((lane & 16) ? 8 : 0);
        out[((size_t)blockIdx.x * 16 + M) * 128 + col] = acc[v] + bias;
    }
}

// ---------------------------------------------------------------------------
extern "C" void kernel_launch(void* const* d_in, const int* in_sizes, int n_in,
                              void* d_out, int out_size, void* d_ws, size_t ws_size,
                              hipStream_t stream) {
    (void)in_sizes; (void)n_in; (void)out_size; (void)ws_size;
    const float* x     = (const float*)d_in[0];
    const int*   eidx  = (const int*)d_in[1];
    const float* ef    = (const float*)d_in[2];
    const int*   etyp  = (const int*)d_in[3];
    const float* lin_w = (const float*)d_in[4];
    const float* lin_b = (const float*)d_in[5];
    const float* bn0_g = (const float*)d_in[6];
    const float* bn0_b = (const float*)d_in[7];
    const float* emb   = (const float*)d_in[8];
    const float* bnetg = (const float*)d_in[9];
    const float* bnetb = (const float*)d_in[10];
    const float* m1_w1 = (const float*)d_in[11];
    const float* m1_b1 = (const float*)d_in[12];
    const float* bn1_g = (const float*)d_in[13];
    const float* bn1_b = (const float*)d_in[14];
    const float* m1_w2 = (const float*)d_in[15];
    const float* m1_b2 = (const float*)d_in[16];
    const float* bn2_g = (const float*)d_in[17];
    const float* bn2_b = (const float*)d_in[18];
    const float* m2_w  = (const float*)d_in[19];
    const float* m2_b  = (const float*)d_in[20];
    const float* bnA_g = (const float*)d_in[21];
    const float* bnA_b = (const float*)d_in[22];
    const float* cv1_w = (const float*)d_in[23];
    const float* cv1_b = (const float*)d_in[24];
    const float* bnB_g = (const float*)d_in[25];
    const float* bnB_b = (const float*)d_in[26];
    const float* cv2_w = (const float*)d_in[27];
    const float* cv2_b = (const float*)d_in[28];
    const float* bnC_g = (const float*)d_in[29];
    const float* bnC_b = (const float*)d_in[30];
    const float* dc_w  = (const float*)d_in[31];
    const float* dc_b  = (const float*)d_in[32];
    const float* out_w = (const float*)d_in[33];
    const float* out_b = (const float*)d_in[34];

    char* base = (char*)d_ws;
    size_t off = 0;
    auto take = [&](size_t bytes) -> char* {
        char* p = base + off;
        off = (off + bytes + 255) & ~(size_t)255;
        return p;
    };
    _Float16* h_pre = (_Float16*)take((size_t)N_ * 32 * 2);
    _Float16* m12   = (_Float16*)take((size_t)NK_ * 32 * 2);
    _Float16* ebufA = (_Float16*)take((size_t)N_ * 96 * 2);
    _Float16* ebufB = (_Float16*)take((size_t)N_ * 96 * 2);
    _Float16* wh    = (_Float16*)take((size_t)33792 * 2);
    float*    stats = (float*)take((size_t)776 * 4);
    float*    coeff = (float*)take((size_t)896 * 4);

    _Float16* w_lin = wh;
    _Float16* w_m1a = wh + 2048;
    _Float16* w_m1b = wh + 3072;
    _Float16* w_m2  = wh + 4096;
    _Float16* w_out = wh + 13312;

    float* st0s = stats;        float* st0q = stats + 32;
    float* st1s = stats + 64;   float* st1q = stats + 96;
    float* st2s = stats + 128;  float* st2q = stats + 160;
    float* stAs = stats + 192;  float* stAq = stats + 288;
    float* stBs = stats + 384;  float* stBq = stats + 480;
    float* stCs = stats + 576;  float* stCq = stats + 672;
    int*   hist = (int*)(stats + 768);

    float* s0 = coeff;        float* t0 = coeff + 32;
    float* s1 = coeff + 64;   float* t1 = coeff + 96;
    float* s2 = coeff + 128;  float* t2 = coeff + 160;
    float* sA = coeff + 192;  float* tA = coeff + 288;
    float* sB = coeff + 384;  float* tB = coeff + 480;
    float* sC = coeff + 576;  float* tC = coeff + 672;
    float* et_tab = coeff + 768;

    k_prep<<<132, 256, 0, stream>>>(stats, wh, lin_w, m1_w1, m1_w2, m2_w, out_w);

    k_node_gemm<<<(NTIL_N + 7) / 8, 256, 0, stream>>>(x, lin_b, w_lin, h_pre, st0s, st0q);
    k_edge_gemm1<<<(NTIL_NK + 7) / 8, 256, 0, stream>>>(ef, m1_b1, w_m1a, m12, st1s, st1q);
    k_hist<<<(NK_ + 255) / 256, 256, 0, stream>>>(etyp, hist);

    k_finalize_bn<<<1, 64, 0, stream>>>(st0s, st0q, bn0_g, bn0_b, s0, t0, 32, 1.f / N_);
    k_finalize_bn<<<1, 64, 0, stream>>>(st1s, st1q, bn1_g, bn1_b, s1, t1, 32, 1.f / NK_);
    k_finalize_et<<<1, 32, 0, stream>>>(hist, emb, bnetg, bnetb, et_tab);

    k_edge_gemm2<<<(NTIL_NK + 7) / 8, 256, 0, stream>>>(m1_b2, w_m1b, s1, t1, m12, st2s, st2q);
    k_finalize_bn<<<1, 64, 0, stream>>>(st2s, st2q, bn2_g, bn2_b, s2, t2, 32, 1.f / NK_);

    k_e_gemm<<<(NTIL_N + 7) / 8, 256, 0, stream>>>(ef, m2_b, w_m2, ebufA, stAs, stAq);
    k_finalize_bn<<<1, 96, 0, stream>>>(stAs, stAq, bnA_g, bnA_b, sA, tA, 81, 1.f / N_);

    int cgrid = (int)(((long)N_ * 81 + 255) / 256);
    k_conv<<<cgrid, 256, 0, stream>>>(ebufA, sA, tA, cv1_w, cv1_b, ebufB, stBs, stBq, 1);
    k_finalize_bn<<<1, 96, 0, stream>>>(stBs, stBq, bnB_g, bnB_b, sB, tB, 81, 1.f / N_);
    k_conv<<<cgrid, 256, 0, stream>>>(ebufB, sB, tB, cv2_w, cv2_b, ebufA, stCs, stCq, 0);
    k_finalize_bn<<<1, 96, 0, stream>>>(stCs, stCq, bnC_g, bnC_b, sC, tC, 81, 1.f / N_);

    k_final<<<NTIL_N, 256, 0, stream>>>(eidx, etyp, h_pre, m12, ebufA,
                                        s0, t0, s2, t2, sC, tC, et_tab,
                                        dc_w, dc_b, w_out, out_b, (float*)d_out);
}
